// MultiheadAttention_27642409517522
// MI455X (gfx1250) — matrix operand order repaired and checked by reading, not compiled, not run on hardware
//
#include <hip/hip_runtime.h>

// ---------------------------------------------------------------------------
// MultiheadAttention for MI455X (gfx1250): bf16 WMMA pipeline
//   B=2, S=2048, E=1024, H=16, Dh=64
// ---------------------------------------------------------------------------

typedef __attribute__((ext_vector_type(16))) __bf16          v16bf;
typedef __attribute__((ext_vector_type(16))) unsigned short  v16us;
typedef __attribute__((ext_vector_type(8)))  unsigned short  v8us;
typedef __attribute__((ext_vector_type(8)))  float           v8f;

#define B_   2
#define S_   2048
#define E_   1024
#define H_   16
#define DH_  64

// fp32 -> bf16, round-to-nearest-even
__device__ __forceinline__ unsigned short f2bf(float f) {
  unsigned u = __float_as_uint(f);
  u += 0x7FFFu + ((u >> 16) & 1u);
  return (unsigned short)(u >> 16);
}

// Fetch a 16x32 bf16 A-fragment for this lane.
// Layout: lane l holds row M=l%16; hi=l/16; K = {16hi..16hi+15} (contiguous).
// p must point at (row_base + k0 + 8*hi); 16-byte aligned.
__device__ __forceinline__ v16us a_frag16(const unsigned short* p) {
  p += 8 * ((threadIdx.x & 31) >> 4);
  v8us lo = *(const v8us*)p;
  v8us hi = *(const v8us*)(p + 8);
  return __builtin_shufflevector(lo, hi, 0,1,2,3,4,5,6,7,8,9,10,11,12,13,14,15);
}

// A-fragment K offset for element e, lane-half hi (used for LDS-sourced A)
__device__ __forceinline__ int a_koff(int e, int hi) {
  return e + (hi << 4);
}

__device__ __forceinline__ v8f wmma_bf16(v16us a_us, v16us b_us, v8f c) {
  v16bf a = __builtin_bit_cast(v16bf, a_us);
  v16bf b = __builtin_bit_cast(v16bf, b_us);
  return __builtin_amdgcn_wmma_f32_16x16x32_bf16(
      /*neg_a=*/false, a, /*neg_b=*/false, b,
      /*c_mod=*/(short)0, c, /*reuse_a=*/false, /*reuse_b=*/false);
}

// ---------------------------------------------------------------------------
// Elementwise helper
// ---------------------------------------------------------------------------
__global__ __launch_bounds__(256) void mha_cvt_f32_bf16(
    const float* __restrict__ src, unsigned short* __restrict__ dst, int n) {
  int i = blockIdx.x * 256 + threadIdx.x;
  if (i < n) dst[i] = f2bf(src[i]);
}

// ---------------------------------------------------------------------------
// QKV projection: qkv[m,n] = sum_k x[m,k] * Win[n,k] + bias[n]
//   M = B*S = 4096, N = 3E = 3072, K = E = 1024
// Block = 128 threads (4 waves). Wave w computes rows [by*64+w*16, +16) x 64 cols.
// Scatter: q scaled by Dh^-0.5 into [B,H,S,Dh]; k into [B,H,S,Dh]; v into [B,H,Dh,S].
// ---------------------------------------------------------------------------
__global__ __launch_bounds__(128) void mha_qkv_gemm(
    const unsigned short* __restrict__ xb,
    const unsigned short* __restrict__ wb,
    const float* __restrict__ bias,
    unsigned short* __restrict__ qb,
    unsigned short* __restrict__ kb,
    unsigned short* __restrict__ vT) {
  const int lane = threadIdx.x & 31;
  const int wv   = threadIdx.x >> 5;
  const int mlo  = lane & 15;
  const int hi   = lane >> 4;
  const int mbase = blockIdx.y * 64 + wv * 16;
  const int nbase = blockIdx.x * 64;

  v8f acc[4];
  #pragma unroll
  for (int t = 0; t < 4; ++t) acc[t] = (v8f){};

  const unsigned short* arow = xb + (size_t)(mbase + mlo) * E_;
  #pragma unroll 4
  for (int kt = 0; kt < E_ / 32; ++kt) {
    const int k0 = kt * 32;
    v16us a = a_frag16(arow + k0 + 8 * hi);
    #pragma unroll
    for (int t = 0; t < 4; ++t) {
      const unsigned short* brow = wb + (size_t)(nbase + t * 16 + mlo) * E_;
      v16us b = *(const v16us*)(brow + k0 + 16 * hi);
      acc[t] = wmma_bf16(a, b, acc[t]);
    }
  }

  #pragma unroll
  for (int t = 0; t < 4; ++t) {
    #pragma unroll
    for (int r = 0; r < 8; ++r) {
      const int m = mbase + hi * 8 + r;        // [0, 4096)
      const int n = nbase + t * 16 + mlo;      // [0, 3072)
      float val = acc[t][r] + bias[n];
      const int bI   = m >> 11;                // batch
      const int srow = m & (S_ - 1);
      const int part = n >> 10;                // 0=q 1=k 2=v
      const int nn   = n & (E_ - 1);
      const int hh   = nn >> 6;
      const int d    = nn & (DH_ - 1);
      if (part == 0) {
        qb[(((size_t)bI * H_ + hh) * S_ + srow) * DH_ + d] = f2bf(val * 0.125f);
      } else if (part == 1) {
        kb[(((size_t)bI * H_ + hh) * S_ + srow) * DH_ + d] = f2bf(val);
      } else {
        vT[(((size_t)bI * H_ + hh) * DH_ + d) * S_ + srow] = f2bf(val);
      }
    }
  }
}

// ---------------------------------------------------------------------------
// Attention core. One block (8 waves) per (b, 16-row query block); loops all H
// heads. LDS: sc[16][2048] fp32 scores (128KB), accum[16][2048] fp32 running
// avg_weights (128KB), pp[1024] fp32 partial-sum staging (4KB) = 260KB/WGP.
// Per head:
//   phase 1: scores = (q*scale) @ k^T via WMMA (8 waves stripe 128 tiles)
//   phase 2: row softmax (fp32, wave reductions); acc += p/H in LDS
//   phase 3: attn = P @ V via WMMA; waves split 4 feature tiles x 2 K-halves,
//            K-half partials combined through pp
// After head loop: single coalesced flush of acc -> avg (no global RMW,
// no atomics, fully deterministic).
// ---------------------------------------------------------------------------
__global__ __launch_bounds__(256) void mha_attention(
    const unsigned short* __restrict__ qb,
    const unsigned short* __restrict__ kb,
    const unsigned short* __restrict__ vT,
    unsigned short* __restrict__ ab,
    float* __restrict__ avg) {
  extern __shared__ float smem[];
  float* sc  = smem;                 // [16][S_]
  float* acc = smem + 16 * S_;       // [16][S_]
  float* pp  = smem + 32 * S_;       // [4][16][16]
  const int tid  = threadIdx.x;
  const int lane = tid & 31;
  const int wv   = tid >> 5;                       // 0..7
  const int mlo  = lane & 15;
  const int hi   = lane >> 4;
  const int qt   = blockIdx.x & ((S_ / 16) - 1);   // query tile
  const int b    = blockIdx.x >> 7;                // S_/16 == 128

  for (int i = tid; i < 16 * S_; i += 256) acc[i] = 0.0f;

  for (int h = 0; h < H_; ++h) {
    const size_t base = ((size_t)b * H_ + h) * (size_t)(S_ * DH_); // q/k/vT share this

    // ---- phase 1: scores ----
    const unsigned short* qrow = qb + base + (size_t)(qt * 16 + mlo) * DH_;
    for (int j = wv; j < S_ / 16; j += 8) {
      v8f c = (v8f){};
      #pragma unroll
      for (int ks = 0; ks < 2; ++ks) {
        v16us a = a_frag16(qrow + ks * 32 + 8 * hi);
        const unsigned short* krow = kb + base + (size_t)(j * 16 + mlo) * DH_;
        v16us bf = *(const v16us*)(krow + ks * 32 + 16 * hi);
        c = wmma_bf16(a, bf, c);
      }
      #pragma unroll
      for (int r = 0; r < 8; ++r)
        sc[(hi * 8 + r) * S_ + j * 16 + mlo] = c[r];
    }
    __syncthreads();

    // ---- phase 2: softmax + LDS avg accumulation (2 rows per wave) ----
    #pragma unroll
    for (int rr = 0; rr < 2; ++rr) {
      const int row = wv * 2 + rr;
      float* scr = sc + row * S_;
      float mx = -3.0e38f;
      for (int col = lane; col < S_; col += 32) mx = fmaxf(mx, scr[col]);
      #pragma unroll
      for (int o = 16; o > 0; o >>= 1) mx = fmaxf(mx, __shfl_xor(mx, o, 32));
      float sum = 0.0f;
      for (int col = lane; col < S_; col += 32) {
        float p = __expf(scr[col] - mx);
        scr[col] = p;
        sum += p;
      }
      #pragma unroll
      for (int o = 16; o > 0; o >>= 1) sum += __shfl_xor(sum, o, 32);
      const float inv = 1.0f / sum;
      float* accr = acc + row * S_;
      for (int col = lane; col < S_; col += 32) {
        float p = scr[col] * inv;
        scr[col] = p;
        accr[col] += p * (1.0f / (float)H_);   // unique (row,col) per thread
      }
    }
    __syncthreads();

    // ---- phase 3: P @ V ; wave = (tile = wv&3, khalf = wv>>2) ----
    const int tile  = wv & 3;
    const int khalf = wv >> 2;
    v8f c = (v8f){};
    const unsigned short* vrow = vT + base + (size_t)(tile * 16 + mlo) * S_;
    for (int kt = khalf * 32; kt < khalf * 32 + 32; ++kt) {
      v16us a;
      #pragma unroll
      for (int e = 0; e < 16; ++e)
        a[e] = f2bf(sc[mlo * S_ + kt * 32 + a_koff(e, hi)]);
      v16us bf = *(const v16us*)(vrow + kt * 32 + 16 * hi);
      c = wmma_bf16(a, bf, c);
    }
    if (khalf == 1) {
      #pragma unroll
      for (int r = 0; r < 8; ++r)
        pp[tile * 256 + (hi * 8 + r) * 16 + mlo] = c[r];
    }
    __syncthreads();
    if (khalf == 0) {
      #pragma unroll
      for (int r = 0; r < 8; ++r) {
        const int m = qt * 16 + hi * 8 + r;
        float o2 = c[r] + pp[tile * 256 + (hi * 8 + r) * 16 + mlo];
        ab[((size_t)(b * S_ + m)) * E_ + h * DH_ + tile * 16 + mlo] = f2bf(o2);
      }
    }
    __syncthreads();   // sc reused next head
  }

  // ---- flush avg_weights (single coalesced write) ----
  for (int i = tid; i < 16 * S_; i += 256) {
    const int row = i >> 11;           // /S_
    const int col = i & (S_ - 1);
    avg[((size_t)b * S_ + qt * 16 + row) * S_ + col] = acc[i];
  }
}

// ---------------------------------------------------------------------------
// Output projection: out[m,n] = sum_k attn[m,k] * Wout[n,k] + bias[n]
//   M = 4096, N = 1024, K = 1024
// ---------------------------------------------------------------------------
__global__ __launch_bounds__(128) void mha_out_gemm(
    const unsigned short* __restrict__ ab,
    const unsigned short* __restrict__ wb,
    const float* __restrict__ bias,
    float* __restrict__ out) {
  const int lane = threadIdx.x & 31;
  const int wv   = threadIdx.x >> 5;
  const int mlo  = lane & 15;
  const int hi   = lane >> 4;
  const int mbase = blockIdx.y * 64 + wv * 16;
  const int nbase = blockIdx.x * 64;

  v8f acc[4];
  #pragma unroll
  for (int t = 0; t < 4; ++t) acc[t] = (v8f){};

  const unsigned short* arow = ab + (size_t)(mbase + mlo) * E_;
  #pragma unroll 4
  for (int kt = 0; kt < E_ / 32; ++kt) {
    const int k0 = kt * 32;
    v16us a = a_frag16(arow + k0 + 8 * hi);
    #pragma unroll
    for (int t = 0; t < 4; ++t) {
      const unsigned short* brow = wb + (size_t)(nbase + t * 16 + mlo) * E_;
      v16us b = *(const v16us*)(brow + k0 + 16 * hi);
      acc[t] = wmma_bf16(a, b, acc[t]);
    }
  }

  #pragma unroll
  for (int t = 0; t < 4; ++t) {
    #pragma unroll
    for (int r = 0; r < 8; ++r) {
      const int m = mbase + hi * 8 + r;
      const int n = nbase + t * 16 + mlo;
      out[(size_t)m * E_ + n] = acc[t][r] + bias[n];
    }
  }
}

// ---------------------------------------------------------------------------
// Launch
// ---------------------------------------------------------------------------
extern "C" void kernel_launch(void* const* d_in, const int* in_sizes, int n_in,
                              void* d_out, int out_size, void* d_ws, size_t ws_size,
                              hipStream_t stream) {
  (void)in_sizes; (void)n_in; (void)out_size; (void)ws_size;

  const float* x     = (const float*)d_in[0];
  const float* w_in  = (const float*)d_in[1];
  const float* b_in  = (const float*)d_in[2];
  const float* w_out = (const float*)d_in[3];
  const float* b_out = (const float*)d_in[4];

  float* out = (float*)d_out;
  float* avg = out + (size_t)B_ * S_ * E_;

  unsigned short* ws = (unsigned short*)d_ws;
  size_t off = 0;
  unsigned short* xb    = ws + off; off += (size_t)B_ * S_ * E_;   // 4M
  unsigned short* winb  = ws + off; off += (size_t)3 * E_ * E_;    // 3M
  unsigned short* woutb = ws + off; off += (size_t)E_ * E_;        // 1M
  unsigned short* qb    = ws + off; off += (size_t)B_ * S_ * E_;   // 4M
  unsigned short* kb    = ws + off; off += (size_t)B_ * S_ * E_;   // 4M
  unsigned short* vT    = ws + off; off += (size_t)B_ * S_ * E_;   // 4M
  unsigned short* ab    = ws + off; off += (size_t)B_ * S_ * E_;   // 4M  (~48MB total)

  const int nx  = B_ * S_ * E_;
  const int nwi = 3 * E_ * E_;
  const int nwo = E_ * E_;

  mha_cvt_f32_bf16<<<(nx  + 255) / 256, 256, 0, stream>>>(x,     xb,    nx);
  mha_cvt_f32_bf16<<<(nwi + 255) / 256, 256, 0, stream>>>(w_in,  winb,  nwi);
  mha_cvt_f32_bf16<<<(nwo + 255) / 256, 256, 0, stream>>>(w_out, woutb, nwo);

  mha_qkv_gemm<<<dim3(3 * E_ / 64, B_ * S_ / 64), 128, 0, stream>>>(
      xb, winb, b_in, qb, kb, vT);

  const size_t attn_lds = (size_t)(32 * S_ + 1024) * sizeof(float); // 260KB
  mha_attention<<<B_ * (S_ / 16), 256, attn_lds, stream>>>(
      qb, kb, vT, ab, avg);

  mha_out_gemm<<<dim3(E_ / 64, B_ * S_ / 64), 128, 0, stream>>>(
      ab, woutb, b_out, out);
}
